// ChebyASPIRELayer_26938034881065
// MI455X (gfx1250) — compile-verified, matrix operand order checked by reading, not run on hardware
//
#include <hip/hip_runtime.h>
#include <hip/hip_bf16.h>

typedef __bf16 bf16_t;
typedef __attribute__((ext_vector_type(16))) __bf16 v16bf;
typedef __attribute__((ext_vector_type(8)))  __bf16 v8bf;
typedef __attribute__((ext_vector_type(8)))  float  v8f;

#define MDIM 8192
#define NDIM 4096
#define BC   256
#define KDEG 20

#define BM 128            // block tile rows (output M)
#define BN 128            // block tile cols (batch)
#define BK 32             // K step = one bf16 WMMA K
#define BS_STRIDE (BK + 8)  // 40 bf16 = 80B row: 16B-aligned chunks, conflict-padded

// ---------------------------------------------------------------------------
// CDNA5 async global->LDS copy (16B per lane), tracked by ASYNCcnt.
// Inline asm per CDNA5 ISA (portable across toolchains).
// ---------------------------------------------------------------------------
__device__ __forceinline__ void async_copy16(const bf16_t* g, bf16_t* l)
{
    unsigned loff = (unsigned)(size_t)l;   // LDS aperture: low 32 bits = LDS byte addr
    asm volatile("global_load_async_to_lds_b128 %0, %1, off"
                 :: "v"(loff), "v"(g)
                 : "memory");
}

template <int N>
__device__ __forceinline__ void wait_asynccnt()
{
    asm volatile("s_wait_asynccnt %0" :: "i"(N) : "memory");
}

// ---------------------------------------------------------------------------
// One-time: X fp32 (M x N) -> Xb bf16 (row-major) and XTb bf16 (transposed).
// Together 128 MB -> resident in the 192 MB L2 for all 40 GEMMs.
// ---------------------------------------------------------------------------
__global__ __launch_bounds__(256) void conv_x_kernel(
    const float* __restrict__ X, bf16_t* __restrict__ Xb, bf16_t* __restrict__ XTb)
{
    size_t id = (size_t)blockIdx.x * blockDim.x + threadIdx.x;
    if (id >= (size_t)MDIM * NDIM) return;
    int m = (int)(id / NDIM);
    int n = (int)(id % NDIM);
    bf16_t b = (bf16_t)X[id];
    Xb[id] = b;
    XTb[(size_t)n * MDIM + m] = b;
}

// Rbf (B x N bf16) = bf16(R); R itself serves as T0 fp32 (state kept B x N).
__global__ __launch_bounds__(256) void conv_r_kernel(
    const float* __restrict__ R, bf16_t* __restrict__ Rbf)
{
    int id = blockIdx.x * blockDim.x + threadIdx.x;
    if (id >= BC * NDIM) return;
    Rbf[id] = (bf16_t)R[id];
}

// ---------------------------------------------------------------------------
// WMMA bf16 GEMM: C(Mdim x BC) = A(Mdim x Kdim bf16 row-major) * B(Kdim x BC)
// where B is supplied TRANSPOSED: Bt (BC x Kdim bf16 row-major) -> staging is a
// pure contiguous async copy into the [col][k] layout WMMA B-fragments want.
//   mode 0: OutBf[colg*out_ld + m] = bf16(C)                      (U^T, B x M)
//   mode 1: T1 = (C - tm*T0)/th ; Acc  = c0*T0 + c1*T1 ; OutBf = bf16(T1)
//   mode 2: Tn = 2*(C - tm*Tc)/th - Tp ; Acc += c_k*Tn ; OutBf = bf16(Tn)
// State arrays (Tprev/Tcur/Tnext/Acc) are (BC x Mdim) fp32 -> contiguous I/O.
// ---------------------------------------------------------------------------
__global__ __launch_bounds__(256) void cheb_wmma_gemm(
    const bf16_t* __restrict__ A, int Kdim,
    const bf16_t* __restrict__ Bt,
    bf16_t*       __restrict__ OutBf, int out_ld,
    const float*  __restrict__ Tprev, const float* __restrict__ Tcur,
    float*        __restrict__ Tnext, float*       __restrict__ Accb,
    const float*  __restrict__ coeffs, int kidx,
    const float*  __restrict__ tmid_p, const float* __restrict__ thalf_p,
    int mode)
{
    __shared__ __attribute__((aligned(16))) bf16_t As[2][BM][BS_STRIDE];
    __shared__ __attribute__((aligned(16))) bf16_t Bs[2][BN][BS_STRIDE];

    const int tid  = threadIdx.x;
    const int lane = tid & 31;
    const int wave = tid >> 5;    // 0..7
    const int wm   = wave >> 2;   // 0..1 : 64-row strip (M)
    const int wn   = wave & 3;    // 0..3 : 32-col strip (batch)
    const int l16  = lane & 15;
    const int hi   = lane >> 4;   // K-half selector of the bf16 fragment layout

    const int blockM = blockIdx.x * BM;
    const int blockN = blockIdx.y * BN;

    // staging map: 128x32 bf16 tile = 512 16B chunks; thread t -> chunks t, t+256
    const int srow = tid >> 2;        // 0..63
    const int skc  = (tid & 3) * 8;   // element offset of 16B chunk within row

    v8f acc[4][2] = {};   // wave tile 64(M) x 32(batch): 8 x (16x16) accumulators

    const int nk = Kdim / BK;

    // ---- prologue: stage tile 0 into buffer 0 (4 async ops / thread) ----
    {
        const int k0 = 0;
        async_copy16(A  + (size_t)(blockM + srow)      * Kdim + k0 + skc, &As[0][srow][skc]);
        async_copy16(A  + (size_t)(blockM + srow + 64) * Kdim + k0 + skc, &As[0][srow + 64][skc]);
        async_copy16(Bt + (size_t)(blockN + srow)      * Kdim + k0 + skc, &Bs[0][srow][skc]);
        async_copy16(Bt + (size_t)(blockN + srow + 64) * Kdim + k0 + skc, &Bs[0][srow + 64][skc]);
    }

    for (int kt = 0; kt < nk; ++kt) {
        const int buf = kt & 1;
        if (kt + 1 < nk) {
            const int k0 = (kt + 1) * BK;
            const int nb = buf ^ 1;
            async_copy16(A  + (size_t)(blockM + srow)      * Kdim + k0 + skc, &As[nb][srow][skc]);
            async_copy16(A  + (size_t)(blockM + srow + 64) * Kdim + k0 + skc, &As[nb][srow + 64][skc]);
            async_copy16(Bt + (size_t)(blockN + srow)      * Kdim + k0 + skc, &Bs[nb][srow][skc]);
            async_copy16(Bt + (size_t)(blockN + srow + 64) * Kdim + k0 + skc, &Bs[nb][srow + 64][skc]);
            wait_asynccnt<4>();   // previous tile's 4 ops done (in-order completion)
        } else {
            wait_asynccnt<0>();
        }
        __syncthreads();

        union F { v16bf v; v8bf h[2]; };
        F af[4], bfr[2];
        #pragma unroll
        for (int tm = 0; tm < 4; ++tm) {
            const bf16_t* p = &As[buf][wm * 64 + tm * 16 + l16][hi * 8];
            af[tm].h[0] = *(const v8bf*)p;          // K = hi*8   .. +7
            af[tm].h[1] = *(const v8bf*)(p + 16);   // K = 16+hi*8.. +7
        }
        #pragma unroll
        for (int tn = 0; tn < 2; ++tn) {
            const bf16_t* p = &Bs[buf][wn * 32 + tn * 16 + l16][hi * 8];
            bfr[tn].h[0] = *(const v8bf*)p;
            bfr[tn].h[1] = *(const v8bf*)(p + 16);
        }
        #pragma unroll
        for (int tm = 0; tm < 4; ++tm)
            #pragma unroll
            for (int tn = 0; tn < 2; ++tn)
                acc[tm][tn] = __builtin_amdgcn_wmma_f32_16x16x32_bf16(
                    false, af[tm].v, false, bfr[tn].v, (short)0, acc[tm][tn], false, false);

        __syncthreads();   // protect buf before it is restaged next iteration
    }

    // ---- epilogue: all outputs contiguous per lane (8 consecutive columns) ----
    const float tmv    = (mode != 0) ? *tmid_p : 0.0f;
    const float inv_th = (mode != 0) ? (1.0f / (*thalf_p)) : 0.0f;
    const float ca     = (mode == 1) ? coeffs[0] : ((mode == 2) ? coeffs[kidx] : 0.0f);
    const float cb     = (mode == 1) ? coeffs[1] : 0.0f;

    #pragma unroll
    for (int tm = 0; tm < 4; ++tm) {
        #pragma unroll
        for (int tn = 0; tn < 2; ++tn) {
            const int row0 = blockM + wm * 64 + tm * 16 + hi * 8;       // output col base
            const int colg = blockN + wn * 32 + tn * 16 + l16;          // output row (batch)
            const size_t base = (size_t)colg * out_ld + row0;
            if (mode == 0) {
                v8bf o;
                #pragma unroll
                for (int i = 0; i < 8; ++i) o[i] = (bf16_t)acc[tm][tn][i];
                *(v8bf*)(OutBf + base) = o;
            } else {
                union V8 { float4 f4[2]; float s[8]; };
                V8 tc, tn8, ac;
                tc.f4[0] = *(const float4*)(Tcur + base);
                tc.f4[1] = *(const float4*)(Tcur + base + 4);
                if (mode == 1) {
                    #pragma unroll
                    for (int i = 0; i < 8; ++i) {
                        tn8.s[i] = (acc[tm][tn][i] - tmv * tc.s[i]) * inv_th;
                        ac.s[i]  = ca * tc.s[i] + cb * tn8.s[i];
                    }
                } else {
                    V8 tp, a0;
                    tp.f4[0] = *(const float4*)(Tprev + base);
                    tp.f4[1] = *(const float4*)(Tprev + base + 4);
                    a0.f4[0] = *(const float4*)(Accb + base);
                    a0.f4[1] = *(const float4*)(Accb + base + 4);
                    #pragma unroll
                    for (int i = 0; i < 8; ++i) {
                        tn8.s[i] = 2.0f * (acc[tm][tn][i] - tmv * tc.s[i]) * inv_th - tp.s[i];
                        ac.s[i]  = a0.s[i] + ca * tn8.s[i];
                    }
                }
                *(float4*)(Tnext + base)     = tn8.f4[0];
                *(float4*)(Tnext + base + 4) = tn8.f4[1];
                *(float4*)(Accb + base)      = ac.f4[0];
                *(float4*)(Accb + base + 4)  = ac.f4[1];
                v8bf o;
                #pragma unroll
                for (int i = 0; i < 8; ++i) o[i] = (bf16_t)tn8.s[i];
                *(v8bf*)(OutBf + base) = o;   // bf16 copy feeds next GEMM1 staging
            }
        }
    }
}

// ---------------------------------------------------------------------------
extern "C" void kernel_launch(void* const* d_in, const int* in_sizes, int n_in,
                              void* d_out, int out_size, void* d_ws, size_t ws_size,
                              hipStream_t stream)
{
    const float* X      = (const float*)d_in[0];   // M x N
    const float* R      = (const float*)d_in[1];   // B x N  (== T0 in B x N layout)
    const float* coeffs = (const float*)d_in[2];   // KDEG+1
    const float* tmid   = (const float*)d_in[3];
    const float* thalf  = (const float*)d_in[4];
    float* Acc = (float*)d_out;                    // B x N : accumulated directly

    // workspace: Xb(64M) | XTb(64M) | Ut(4M) | Rbf(2M) | Tbf(2M) | T fp32 x3 (12M)
    char* ws = (char*)d_ws;
    bf16_t* Xb  = (bf16_t*)ws;
    bf16_t* XTb = (bf16_t*)(ws + (size_t)MDIM * NDIM * sizeof(bf16_t));
    size_t off  = (size_t)MDIM * NDIM * 2 * sizeof(bf16_t);
    bf16_t* Ut  = (bf16_t*)(ws + off); off += (size_t)BC * MDIM * sizeof(bf16_t);
    bf16_t* Rbf = (bf16_t*)(ws + off); off += (size_t)BC * NDIM * sizeof(bf16_t);
    bf16_t* Tbf = (bf16_t*)(ws + off); off += (size_t)BC * NDIM * sizeof(bf16_t);
    float* Tf[3];
    for (int i = 0; i < 3; ++i) { Tf[i] = (float*)(ws + off); off += (size_t)BC * NDIM * sizeof(float); }

    conv_x_kernel<<<(unsigned)(((size_t)MDIM * NDIM) / 256), 256, 0, stream>>>(X, Xb, XTb);
    conv_r_kernel<<<(BC * NDIM) / 256, 256, 0, stream>>>(R, Rbf);

    dim3 g1(MDIM / BM, BC / BN);   // 64 x 2 blocks : U^T = (X * T)^T
    dim3 g2(NDIM / BM, BC / BN);   // 32 x 2 blocks : W   = X^T * U  (+ cheb epilogue)

    // iteration 1: T1 = z_scaled(T0); Acc = c0*T0 + c1*T1   (T0 fp32 == R)
    cheb_wmma_gemm<<<g1, 256, 0, stream>>>(Xb, NDIM, Rbf, Ut, MDIM,
                                           nullptr, nullptr, nullptr, nullptr,
                                           coeffs, 0, tmid, thalf, 0);
    cheb_wmma_gemm<<<g2, 256, 0, stream>>>(XTb, MDIM, Ut, Tbf, NDIM,
                                           nullptr, R, Tf[0], Acc,
                                           coeffs, 1, tmid, thalf, 1);

    const float* tprev = R;
    const float* tcur  = Tf[0];
    float*       tnext = Tf[1];
    float*       spare = Tf[2];
    for (int k = 2; k <= KDEG; ++k) {
        cheb_wmma_gemm<<<g1, 256, 0, stream>>>(Xb, NDIM, Tbf, Ut, MDIM,
                                               nullptr, nullptr, nullptr, nullptr,
                                               coeffs, 0, tmid, thalf, 0);
        cheb_wmma_gemm<<<g2, 256, 0, stream>>>(XTb, MDIM, Ut, Tbf, NDIM,
                                               tprev, tcur, tnext, Acc,
                                               coeffs, k, tmid, thalf, 2);
        float* newnext = (k == 2) ? spare : (float*)tprev;  // R never overwritten
        tprev = tcur;
        tcur  = tnext;
        tnext = newnext;
    }
}
